// MultiHeadAttention_79173427135001
// MI455X (gfx1250) — compile-verified, matrix operand order checked by reading, not run on hardware
//
#include <hip/hip_runtime.h>

typedef __bf16 bf16;
typedef __attribute__((ext_vector_type(16))) __bf16 v16bf;
typedef __attribute__((ext_vector_type(8)))  __bf16 v8bf;
typedef __attribute__((ext_vector_type(4)))  __bf16 v4bf;
typedef __attribute__((ext_vector_type(8)))  float  v8f;

#define D_MODEL 1024
#define T_SEQ   2048
#define NHEADS  16
#define HDIM    64
#define BROWS   4096   // B*T

// ---------------------------------------------------------------------------
// WMMA wrapper: D = A(16x32 bf16) * B(32x16 bf16) + C (f32)
// ---------------------------------------------------------------------------
__device__ inline v8f wmma_bf16(v16bf a, v16bf b, v8f c) {
    return __builtin_amdgcn_wmma_f32_16x16x32_bf16(
        /*neg_a=*/false, a, /*neg_b=*/false, b,
        /*c_mod=*/(short)0, c, /*reuse_a=*/false, /*reuse_b=*/false);
}

// Load one 16-element bf16 fragment for this lane (global or generic ptr).
// ISA 16-bit A/B layout: elems 0..7 -> K = half*8 + j (at `p`),
//                        elems 8..15 -> K = 16 + half*8 + j (at `p+16`).
__device__ inline v16bf load_frag(const bf16* p) {
    v8bf lo = *(const v8bf*)(p);
    v8bf hi = *(const v8bf*)(p + 16);
    return __builtin_shufflevector(lo, hi, 0,1,2,3,4,5,6,7,8,9,10,11,12,13,14,15);
}

// Async copy of 16 bytes global -> LDS (CDNA5 GLOBAL_LOAD_ASYNC_TO_LDS_B128,
// tracked by ASYNCcnt). ldsoff = byte offset in LDS, gptr = 64-bit global addr.
__device__ inline void async_g2l_b128(unsigned ldsoff, const void* gptr) {
    asm volatile("global_load_async_to_lds_b128 %0, %1, off"
                 :: "v"(ldsoff), "v"((unsigned long long)(uintptr_t)gptr)
                 : "memory");
}
__device__ inline void wait_asynccnt0() {
    asm volatile("s_wait_asynccnt 0x0" ::: "memory");
}

// ---------------------------------------------------------------------------
// fp32 -> bf16 conversion (vectorized x4)
// ---------------------------------------------------------------------------
__global__ void cvt_f32_bf16(const float* __restrict__ in, bf16* __restrict__ out, int n4) {
    int i = blockIdx.x * blockDim.x + threadIdx.x;
    if (i < n4) {
        float4 f = ((const float4*)in)[i];
        v4bf r;
        r[0] = (bf16)f.x; r[1] = (bf16)f.y; r[2] = (bf16)f.z; r[3] = (bf16)f.w;
        ((v4bf*)out)[i] = r;
    }
}

// ---------------------------------------------------------------------------
// GEMM: C[m,n] = sum_k A[m,k] * W[n,k]   (A: 4096x1024 bf16, W: 1024x1024 bf16)
// Block = 4 waves sharing one 64-column weight slice (nt), each wave a 32-row
// tile -> block computes 128x64. Per 32-deep k-stage the whole block async-
// copies the 64x32 B tile into LDS (2 stages, double-buffered, 8KB total);
// each wave reads B fragments from LDS (4x less global weight traffic) and
// keeps its 2 A fragments register-double-buffered from global.
// Ordering: own-wave s_wait_asynccnt 0, then one barrier per stage; next-
// buffer async writes are issued only after the barrier that follows the
// previous reads of that buffer.
// MODE 0: store f32 row-major (final output)
// MODE 1: store bf16 to head layout [b, h, t, 64]        (Q, K)
// MODE 2: store bf16 to transposed head layout [b, h, 64, t]  (V^T)
// ---------------------------------------------------------------------------
template <int MODE>
__global__ __launch_bounds__(128) void gemm_tile(const bf16* __restrict__ A,
                                                 const bf16* __restrict__ W,
                                                 void* __restrict__ out) {
    __shared__ __align__(16) bf16 Bsm[2][64 * 32];   // [stage][row*32 + col]

    const int tid  = threadIdx.x;
    const int lane = tid & 31;
    const int wave = tid >> 5;
    const int nt   = blockIdx.x & 15;          // 16 column super-tiles (64 wide)
    const int mtg  = blockIdx.x >> 4;          // 32 row groups
    const int mt   = mtg * 4 + wave;           // 128 row tiles (32 rows each)
    const int half = lane >> 4;
    const int r16  = lane & 15;

    // B-tile async copy: thread -> (row = tid/2, 16-col half = tid&1)
    const int brow  = tid >> 1;
    const int bcol0 = (tid & 1) * 16;
    const bf16* bsrc = W + (size_t)(nt * 64 + brow) * D_MODEL + bcol0;
    const unsigned lds0 = (unsigned)(uintptr_t)&Bsm[0][brow * 32 + bcol0];
    const unsigned lds1 = (unsigned)(uintptr_t)&Bsm[1][brow * 32 + bcol0];

    const bf16* abase = A + (size_t)(mt * 32 + r16) * D_MODEL + half * 8;

    v8f acc[2][4];
    v8f zero = {};
#pragma unroll
    for (int i = 0; i < 2; ++i)
#pragma unroll
        for (int j = 0; j < 4; ++j) acc[i][j] = zero;

    // prologue: stage 0 of B into LDS, A fragments into registers
    async_g2l_b128(lds0, bsrc);
    async_g2l_b128(lds0 + 16, bsrc + 8);

    v16bf af[2][2];
#pragma unroll
    for (int i = 0; i < 2; ++i)
        af[0][i] = load_frag(abase + (size_t)(i * 16) * D_MODEL);

    for (int kk = 0; kk < D_MODEL; kk += 32) {
        const int cur = (kk >> 5) & 1;
        const int nxt = cur ^ 1;

        wait_asynccnt0();        // own writes of stage `cur` complete
        __syncthreads();         // all waves' writes complete; prev reads done

        if (kk + 32 < D_MODEL) {
            const bf16* bs = bsrc + kk + 32;
            const unsigned ld = cur ? lds0 : lds1;
            async_g2l_b128(ld, bs);
            async_g2l_b128(ld + 16, bs + 8);
#pragma unroll
            for (int i = 0; i < 2; ++i)
                af[nxt][i] = load_frag(abase + (size_t)(i * 16) * D_MODEL + kk + 32);
        }

        // B fragments from LDS (ds_load_b128 pairs), then 8 WMMAs
#pragma unroll
        for (int j = 0; j < 4; ++j) {
            const bf16* bp = &Bsm[cur][(j * 16 + r16) * 32 + half * 8];
            v16bf bfj = load_frag(bp);
#pragma unroll
            for (int i = 0; i < 2; ++i)
                acc[i][j] = wmma_bf16(af[cur][i], bfj, acc[i][j]);
        }
    }

    // D layout: lane holds column n, VGPR r holds row m = r + 8*half.
#pragma unroll
    for (int i = 0; i < 2; ++i) {
#pragma unroll
        for (int j = 0; j < 4; ++j) {
            const int n = nt * 64 + j * 16 + r16;
#pragma unroll
            for (int r = 0; r < 8; ++r) {
                const int Mg = mt * 32 + i * 16 + r + 8 * half;   // row in [0,4096)
                if constexpr (MODE == 0) {
                    ((float*)out)[(size_t)Mg * D_MODEL + n] = acc[i][j][r];
                } else {
                    const int bb = Mg >> 11;                      // batch
                    const int t  = Mg & (T_SEQ - 1);
                    const int h  = n >> 6;
                    const int dh = n & (HDIM - 1);
                    bf16* O = (bf16*)out;
                    if constexpr (MODE == 1)
                        O[(((size_t)(bb * NHEADS + h)) * T_SEQ + t) * HDIM + dh] =
                            (bf16)acc[i][j][r];
                    else
                        O[(((size_t)(bb * NHEADS + h)) * HDIM + dh) * T_SEQ + t] =
                            (bf16)acc[i][j][r];
                }
            }
        }
    }
}

// ---------------------------------------------------------------------------
// Flash attention, one wave per (b, h, 32-query block) = two 16-query tiles
// sharing all K and V fragments. Computes S^T = K * Q^T so that:
//   * softmax per query = in-lane reduce over 8 VGPRs + shfl_xor(16)
//   * bf16(P) lands directly in the A-fragment layout for P*V (no shuffles)
// V loads issue before the softmax VALU block; next chunk's K/V lines are
// prefetched (global_prefetch_b8) at chunk start.
// Q,K: [b,h,t,64] bf16;  Vt: [b,h,64,t] bf16;  Ob: [b,t,1024] bf16
// ---------------------------------------------------------------------------
__global__ __launch_bounds__(128) void attn_kernel(const bf16* __restrict__ Q,
                                                   const bf16* __restrict__ K,
                                                   const bf16* __restrict__ Vt,
                                                   bf16* __restrict__ Ob) {
    const int lane = threadIdx.x & 31;
    const int wid  = blockIdx.x * (blockDim.x >> 5) + (threadIdx.x >> 5);
    const int QB   = T_SEQ / 32;                  // 64 query blocks per head
    const int qt   = wid & (QB - 1);
    const int h    = (wid / QB) & (NHEADS - 1);
    const int bb   = wid / (QB * NHEADS);
    const int half = lane >> 4;
    const int q16  = lane & 15;

    const size_t hoff = ((size_t)(bb * NHEADS + h)) * T_SEQ * HDIM;
    const bf16* Qh = Q  + hoff;
    const bf16* Kh = K  + hoff;
    const bf16* Vh = Vt + hoff;                   // [64][2048]

    const int qbase = qt * 32;

    // Q B-fragments for both q-tiles: [qtile][dchunk]
    v16bf qb[2][2];
#pragma unroll
    for (int u = 0; u < 2; ++u) {
        const int qg = qbase + u * 16 + q16;
        qb[u][0] = load_frag(Qh + (size_t)qg * HDIM + half * 8);
        qb[u][1] = load_frag(Qh + (size_t)qg * HDIM + 32 + half * 8);
    }

    v8f   ov[2][4];
    v8f   zero = {};
#pragma unroll
    for (int u = 0; u < 2; ++u)
#pragma unroll
        for (int dt = 0; dt < 4; ++dt) ov[u][dt] = zero;
    float m_run[2] = {-1e30f, -1e30f};
    float s_run[2] = {0.0f, 0.0f};

    for (int kb = 0; kb <= qbase + 31; kb += 32) {
        // prefetch next chunk's K rows and V lines into cache
        if (kb + 32 <= qbase + 31) {
            __builtin_prefetch(Kh + (size_t)(kb + 32 + q16) * HDIM, 0, 0);
            __builtin_prefetch(Kh + (size_t)(kb + 48 + q16) * HDIM, 0, 0);
            __builtin_prefetch(Vh + (size_t)((lane & 3) * 16 + (lane >> 2)) * T_SEQ + kb + 32, 0, 0);
        }

        // --- S^T tiles; each K fragment pair is consumed immediately ---
        v8f s[2][2];
#pragma unroll
        for (int kt = 0; kt < 2; ++kt) {
            const bf16* kp = Kh + (size_t)(kb + kt * 16 + q16) * HDIM + half * 8;
            v16bf ka0 = load_frag(kp);
            v16bf ka1 = load_frag(kp + 32);
#pragma unroll
            for (int u = 0; u < 2; ++u) {
                v8f t0 = wmma_bf16(ka0, qb[u][0], zero);
                s[u][kt] = wmma_bf16(ka1, qb[u][1], t0);
            }
        }

        // --- V B-fragments (shared by q-tiles); issue before softmax VALU ---
        v16bf vb[4];
#pragma unroll
        for (int dt = 0; dt < 4; ++dt)
            vb[dt] = load_frag(Vh + (size_t)(dt * 16 + q16) * T_SEQ + kb + half * 8);

        // --- softmax per q-tile (column q = lane%16 of the S^T tiles) ---
        v16bf pa[2];
        float alpha[2];
#pragma unroll
        for (int u = 0; u < 2; ++u) {
            const int qg = qbase + u * 16 + q16;
            float mtile = -1e30f;
#pragma unroll
            for (int r = 0; r < 8; ++r) {
                const int kg0 = kb + r + 8 * half;
                const int kg1 = kg0 + 16;
                s[u][0][r] = (kg0 <= qg) ? s[u][0][r] * 0.125f : -1e30f;
                s[u][1][r] = (kg1 <= qg) ? s[u][1][r] * 0.125f : -1e30f;
                mtile = fmaxf(mtile, fmaxf(s[u][0][r], s[u][1][r]));
            }
            mtile = fmaxf(mtile, __shfl_xor(mtile, 16, 32));
            const float mnew = fmaxf(m_run[u], mtile);
            alpha[u] = __expf(m_run[u] - mnew);
            float ssum = 0.0f;
#pragma unroll
            for (int r = 0; r < 8; ++r) {
                const float p0 = __expf(s[u][0][r] - mnew);
                const float p1 = __expf(s[u][1][r] - mnew);
                ssum += p0 + p1;
                pa[u][r]     = (bf16)p0;    // A-frag elems 0..7  -> keys kb..kb+15
                pa[u][r + 8] = (bf16)p1;    // A-frag elems 8..15 -> keys kb+16..kb+31
            }
            ssum += __shfl_xor(ssum, 16, 32);
            s_run[u] = s_run[u] * alpha[u] + ssum;
            m_run[u] = mnew;
        }

        // --- rescale O rows (row q = r + 8*half lives in lane q of stats) ---
#pragma unroll
        for (int u = 0; u < 2; ++u)
#pragma unroll
            for (int r = 0; r < 8; ++r) {
                const float ar = __shfl(alpha[u], r + 8 * half, 32);
                ov[u][0][r] *= ar; ov[u][1][r] *= ar;
                ov[u][2][r] *= ar; ov[u][3][r] *= ar;
            }

        // --- O += P * V ---
#pragma unroll
        for (int u = 0; u < 2; ++u)
#pragma unroll
            for (int dt = 0; dt < 4; ++dt)
                ov[u][dt] = wmma_bf16(pa[u], vb[dt], ov[u][dt]);
    }

    // --- normalize and store: Ob[b, t, h*64 + d] bf16 ---
#pragma unroll
    for (int u = 0; u < 2; ++u)
#pragma unroll
        for (int r = 0; r < 8; ++r) {
            const float sr  = __shfl(s_run[u], r + 8 * half, 32);
            const float inv = 1.0f / sr;
            const int   qr  = qbase + u * 16 + r + 8 * half;
            bf16* dst = Ob + ((size_t)bb * T_SEQ + qr) * D_MODEL + h * HDIM;
#pragma unroll
            for (int dt = 0; dt < 4; ++dt)
                dst[dt * 16 + q16] = (bf16)(ov[u][dt][r] * inv);
        }
}

// ---------------------------------------------------------------------------
extern "C" void kernel_launch(void* const* d_in, const int* in_sizes, int n_in,
                              void* d_out, int out_size, void* d_ws, size_t ws_size,
                              hipStream_t stream) {
    const float* x  = (const float*)d_in[0];
    const float* Wq = (const float*)d_in[1];
    const float* Wk = (const float*)d_in[2];
    const float* Wv = (const float*)d_in[3];
    const float* Wo = (const float*)d_in[4];
    // d_in[5] = attn_mask (causal; encoded directly in the kernel)

    const size_t NX = (size_t)BROWS * D_MODEL;       // 4M elems
    const size_t NW = (size_t)D_MODEL * D_MODEL;     // 1M elems

    bf16* xb  = (bf16*)d_ws;
    bf16* wqb = xb  + NX;
    bf16* wkb = wqb + NW;
    bf16* wvb = wkb + NW;
    bf16* wob = wvb + NW;
    bf16* Qh  = wob + NW;       // [b,h,t,64]
    bf16* Kh  = Qh  + NX;       // [b,h,t,64]
    bf16* Vt  = Kh  + NX;       // [b,h,64,t]
    bf16* Obf = Vt  + NX;       // [b,t,1024]

    // 1) convert inputs to bf16
    cvt_f32_bf16<<<(int)(NX / 4 / 256), 256, 0, stream>>>(x,  xb,  (int)(NX / 4));
    cvt_f32_bf16<<<(int)(NW / 4 / 256), 256, 0, stream>>>(Wq, wqb, (int)(NW / 4));
    cvt_f32_bf16<<<(int)(NW / 4 / 256), 256, 0, stream>>>(Wk, wkb, (int)(NW / 4));
    cvt_f32_bf16<<<(int)(NW / 4 / 256), 256, 0, stream>>>(Wv, wvb, (int)(NW / 4));
    cvt_f32_bf16<<<(int)(NW / 4 / 256), 256, 0, stream>>>(Wo, wob, (int)(NW / 4));

    // 2) projections: blocks = 16 nt x 32 row-groups = 512, 128 threads each
    gemm_tile<1><<<512, 128, 0, stream>>>(xb, wqb, Qh);
    gemm_tile<1><<<512, 128, 0, stream>>>(xb, wkb, Kh);
    gemm_tile<2><<<512, 128, 0, stream>>>(xb, wvb, Vt);

    // 3) attention: 2*16*64 = 2048 waves -> 512 blocks of 128 threads
    attn_kernel<<<512, 128, 0, stream>>>(Qh, Kh, Vt, Obf);

    // 4) output projection, f32 result straight to d_out
    gemm_tile<0><<<512, 128, 0, stream>>>(Obf, wob, d_out);
}
